// EncoderLayer_86354612453828
// MI455X (gfx1250) — matrix operand order repaired and checked by reading, not compiled, not run on hardware
//
#include <hip/hip_runtime.h>
#include <hip/hip_bf16.h>
#include <cstdint>

typedef __attribute__((ext_vector_type(16))) _Float16 v16h;
typedef __attribute__((ext_vector_type(8)))  float    v8f;

#define CDIM   512
#define HEADS  8
#define HIDDEN 1024

// ---------- monotone float<->uint mapping for atomicMax-based segment max ----------
__device__ __forceinline__ unsigned f2mono(float f) {
  unsigned b = __float_as_uint(f);
  return (b & 0x80000000u) ? ~b : (b | 0x80000000u);
}
__device__ __forceinline__ float mono2f(unsigned u) {
  unsigned b = (u & 0x80000000u) ? (u ^ 0x80000000u) : ~u;
  return __uint_as_float(b);
}

// ---------- LayerNorm (f32 in) -> f16 out, one row (C=512) per 256-thread block ----------
__global__ void __launch_bounds__(256)
ln_to_f16_kernel(const float* __restrict__ x, const float* __restrict__ g,
                 const float* __restrict__ b, _Float16* __restrict__ z) {
  __shared__ float ssum[256], ssq[256];
  int row = blockIdx.x, t = threadIdx.x;
  const float* xr = x + (size_t)row * CDIM;
  float v0 = xr[t], v1 = xr[t + 256];
  ssum[t] = v0 + v1;
  ssq[t]  = v0 * v0 + v1 * v1;
  __syncthreads();
  for (int o = 128; o > 0; o >>= 1) {
    if (t < o) { ssum[t] += ssum[t + o]; ssq[t] += ssq[t + o]; }
    __syncthreads();
  }
  float mu  = ssum[0] * (1.0f / CDIM);
  float var = ssq[0] * (1.0f / CDIM) - mu * mu;
  float rs  = rsqrtf(var + 1e-5f);
  _Float16* zr = z + (size_t)row * CDIM;
  zr[t]       = (_Float16)((v0 - mu) * rs * g[t]       + b[t]);
  zr[t + 256] = (_Float16)((v1 - mu) * rs * g[t + 256] + b[t + 256]);
}

// ---------- W (N x K, f32, torch Linear layout) -> Wt (K x N, f16) ----------
__global__ void w_transpose_f16_kernel(const float* __restrict__ W,
                                       _Float16* __restrict__ Wt, int N, int K) {
  int i = blockIdx.x * blockDim.x + threadIdx.x;
  if (i >= N * K) return;
  int n = i / K, k = i - n * K;
  Wt[(size_t)k * N + n] = (_Float16)W[i];
}

__global__ void f32_to_f16_kernel(const float* __restrict__ s, _Float16* __restrict__ d, int n) {
  int i = blockIdx.x * blockDim.x + threadIdx.x;
  if (i < n) d[i] = (_Float16)s[i];
}

__global__ void attn_init_kernel(unsigned* __restrict__ mmax, float* __restrict__ denom,
                                 float* __restrict__ agg, int LH, int LC) {
  int i = blockIdx.x * blockDim.x + threadIdx.x;
  if (i < LH) { mmax[i] = 0u; denom[i] = 0.f; }
  if (i < LC) agg[i] = 0.f;
}

// ---------- WMMA fragment loads per documented CDNA5 16-bit layouts ----------
// A 16x32 f16 (row-major source, stride ld):
//   lanes 0-15: M=lane, VGPR i(0-3): K=2i,2i+1 ; VGPR i+4: K=16+2i
//   lanes 16-31: same M, K offset +8
__device__ __forceinline__ v16h load_a_frag(const _Float16* __restrict__ A,
                                            int m0, int k0, int ld, int lane) {
  const _Float16* r = A + (size_t)(m0 + (lane & 15)) * ld + k0 + ((lane >> 4) << 3);
  v16h a;
#pragma unroll
  for (int i = 0; i < 4; ++i) {
    a[2 * i]     = r[2 * i];
    a[2 * i + 1] = r[2 * i + 1];
    a[2 * i + 8] = r[16 + 2 * i];
    a[2 * i + 9] = r[16 + 2 * i + 1];
  }
  return a;
}

// B 32x16 f16 from K-major Wt (K x N, stride ld): N = lane%16, element i(0-7): K=i, i+8: K=16+i; lanes 16-31 K offset +8
__device__ __forceinline__ v16h load_b_frag(const _Float16* __restrict__ Bt,
                                            int k0, int n0, int ld, int lane) {
  const _Float16* r = Bt + (size_t)(k0 + ((lane >> 4) << 3)) * ld + n0 + (lane & 15);
  v16h b;
#pragma unroll
  for (int i = 0; i < 8; ++i) { b[i] = r[(size_t)i * ld]; b[i + 8] = r[(size_t)(16 + i) * ld]; }
  return b;
}

// ---------- GEMM: out(M,N) = A(M,K)f16 @ Wt(K,N)f16 + bias, f32 accum ----------
// EPI: 0 = store f32 ; 1 = residual-add then store f32 ; 2 = SiLU then store f16
template <int EPI>
__global__ void __launch_bounds__(128)
gemm_f16_wmma_kernel(const _Float16* __restrict__ A, const _Float16* __restrict__ Bt,
                     const float* __restrict__ bias, const float* __restrict__ res,
                     float* __restrict__ outF, _Float16* __restrict__ outH,
                     int M, int N, int K) {
  int lane = threadIdx.x & 31;
  int wave = threadIdx.x >> 5;                 // 4 waves per block
  int m0 = blockIdx.y * 64 + (wave >> 1) * 32; // each wave: 32x32 output tile
  int n0 = blockIdx.x * 64 + (wave & 1) * 32;
  if (m0 >= M) return;                         // M is a multiple of 32 here

  v8f acc00, acc01, acc10, acc11;
#pragma unroll
  for (int r = 0; r < 8; ++r) { acc00[r] = 0.f; acc01[r] = 0.f; acc10[r] = 0.f; acc11[r] = 0.f; }

  for (int k0 = 0; k0 < K; k0 += 32) {
    v16h a0 = load_a_frag(A, m0,      k0, K, lane);
    v16h a1 = load_a_frag(A, m0 + 16, k0, K, lane);
    v16h b0 = load_b_frag(Bt, k0, n0,      N, lane);
    v16h b1 = load_b_frag(Bt, k0, n0 + 16, N, lane);
    if (k0 + 32 < K) {
      __builtin_prefetch(A  + (size_t)(m0 + (lane & 15)) * K + k0 + 32, 0, 1);
      __builtin_prefetch(Bt + (size_t)(k0 + 32 + lane) * N + n0,        0, 1);
    }
    acc00 = __builtin_amdgcn_wmma_f32_16x16x32_f16(false, a0, false, b0, (short)0, acc00, false, false);
    acc01 = __builtin_amdgcn_wmma_f32_16x16x32_f16(false, a0, false, b1, (short)0, acc01, false, false);
    acc10 = __builtin_amdgcn_wmma_f32_16x16x32_f16(false, a1, false, b0, (short)0, acc10, false, false);
    acc11 = __builtin_amdgcn_wmma_f32_16x16x32_f16(false, a1, false, b1, (short)0, acc11, false, false);
  }

  int nl = lane & 15;
  int mo = (lane >> 4) << 3; // lanes 16-31 hold M rows +8

#define EPILOGUE(ACC, MI, NI)                                                \
  do {                                                                       \
    int mbase = m0 + (MI) * 16 + mo;                                         \
    int nn = n0 + (NI) * 16 + nl;                                            \
    float bb = bias[nn];                                                     \
    _Pragma("unroll") for (int r = 0; r < 8; ++r) {                          \
      size_t idx = (size_t)(mbase + r) * N + nn;                             \
      float vv = ACC[r] + bb;                                                \
      if (EPI == 0) outF[idx] = vv;                                          \
      else if (EPI == 1) outF[idx] = res[idx] + vv;                          \
      else { float sg = vv / (1.0f + __expf(-vv)); outH[idx] = (_Float16)sg; } \
    }                                                                        \
  } while (0)

  EPILOGUE(acc00, 0, 0);
  EPILOGUE(acc01, 0, 1);
  EPILOGUE(acc10, 1, 0);
  EPILOGUE(acc11, 1, 1);
#undef EPILOGUE
}

// ---------- edge attention: one wave per edge (lane*16 floats = full 512-row) ----------
__global__ void __launch_bounds__(256)
edge_score_kernel(const float* __restrict__ q, const float* __restrict__ k,
                  const long long* __restrict__ ei, const float* __restrict__ bias,
                  float* __restrict__ s, unsigned* __restrict__ mmax, int E) {
  int e = blockIdx.x * 8 + (threadIdx.x >> 5);
  if (e >= E) return;
  int lane = threadIdx.x & 31;
  long long sn = ei[e];
  long long dn = ei[(size_t)E + e];
  const float4* qr = (const float4*)(q + (size_t)sn * CDIM) + lane * 4;
  const float4* kr = (const float4*)(k + (size_t)dn * CDIM) + lane * 4;
  float part = 0.f;
#pragma unroll
  for (int j = 0; j < 4; ++j) {
    float4 a = qr[j], b = kr[j];
    part += a.x * b.x + a.y * b.y + a.z * b.z + a.w * b.w;
  }
  // head h = lane>>2 spans 4 lanes; reduce within the quad
  part += __shfl_xor(part, 1, 32);
  part += __shfl_xor(part, 2, 32);
  if ((lane & 3) == 0) {
    int h = lane >> 2;
    float sv = part * 0.125f + bias[(size_t)e * HEADS + h]; // 1/sqrt(64)
    s[(size_t)e * HEADS + h] = sv;
    atomicMax(&mmax[(size_t)sn * HEADS + h], f2mono(sv));
  }
}

__global__ void edge_p_kernel(const float* __restrict__ s, const long long* __restrict__ ei,
                              const unsigned* __restrict__ mmax, float* __restrict__ p,
                              float* __restrict__ denom, int E) {
  int i = blockIdx.x * blockDim.x + threadIdx.x;
  if (i >= E * HEADS) return;
  int e = i >> 3, h = i & 7;
  long long sn = ei[e];
  float m = mono2f(mmax[(size_t)sn * HEADS + h]);
  float pv = __expf(s[i] - m);
  p[i] = pv;
  atomicAdd(&denom[(size_t)sn * HEADS + h], pv);
}

__global__ void __launch_bounds__(256)
edge_agg_kernel(const float* __restrict__ p, const float* __restrict__ denom,
                const float* __restrict__ v, const long long* __restrict__ ei,
                float* __restrict__ agg, int E) {
  int e = blockIdx.x * 8 + (threadIdx.x >> 5);
  if (e >= E) return;
  int lane = threadIdx.x & 31;
  long long sn = ei[e];
  long long dn = ei[(size_t)E + e];
  int h = lane >> 2;
  float d = denom[(size_t)sn * HEADS + h];
  float alpha = p[(size_t)e * HEADS + h] / (d == 0.f ? 1.f : d);
  const float4* vr = (const float4*)(v + (size_t)dn * CDIM) + lane * 4;
  float* ar = agg + (size_t)sn * CDIM + lane * 16;
#pragma unroll
  for (int j = 0; j < 4; ++j) {
    float4 vj = vr[j];
    atomicAdd(&ar[4 * j + 0], alpha * vj.x);
    atomicAdd(&ar[4 * j + 1], alpha * vj.y);
    atomicAdd(&ar[4 * j + 2], alpha * vj.z);
    atomicAdd(&ar[4 * j + 3], alpha * vj.w);
  }
}

extern "C" void kernel_launch(void* const* d_in, const int* in_sizes, int n_in,
                              void* d_out, int out_size, void* d_ws, size_t ws_size,
                              hipStream_t stream) {
  const float*     x   = (const float*)d_in[0];
  const long long* ei  = (const long long*)d_in[1];  // int64 edge_index (2,E)
  const float*     ab  = (const float*)d_in[2];
  const float* Wq = (const float*)d_in[3];  const float* bq = (const float*)d_in[4];
  const float* Wk = (const float*)d_in[5];  const float* bk = (const float*)d_in[6];
  const float* Wv = (const float*)d_in[7];  const float* bv = (const float*)d_in[8];
  const float* Wo = (const float*)d_in[9];  const float* bo = (const float*)d_in[10];
  const float* lag = (const float*)d_in[11]; const float* lab = (const float*)d_in[12];
  const float* lfg = (const float*)d_in[13]; const float* lfb = (const float*)d_in[14];
  const float* W1 = (const float*)d_in[15]; const float* b1 = (const float*)d_in[16];
  const float* W2 = (const float*)d_in[17]; const float* b2 = (const float*)d_in[18];

  const int L = in_sizes[0] / CDIM;
  const int E = in_sizes[1] / 2;
  float* out = (float*)d_out;

  // ---- carve workspace ----
  char* ws = (char*)d_ws;
  size_t off = 0;
  auto alloc = [&](size_t bytes) -> void* {
    void* ptr = ws + off;
    off += (bytes + 255) & ~(size_t)255;
    return ptr;
  };
  _Float16* zh   = (_Float16*)alloc((size_t)L * CDIM * 2);  // LN1 out -> agg f16 -> LN2 out (serial reuse)
  float*    q    = (float*)alloc((size_t)L * CDIM * 4);
  float*    kbuf = (float*)alloc((size_t)L * CDIM * 4);
  float*    vbuf = (float*)alloc((size_t)L * CDIM * 4);
  float*    s    = (float*)alloc((size_t)E * HEADS * 4);
  float*    p    = (float*)alloc((size_t)E * HEADS * 4);
  unsigned* mmax = (unsigned*)alloc((size_t)L * HEADS * 4);
  float*    den  = (float*)alloc((size_t)L * HEADS * 4);
  float*    agg  = (float*)alloc((size_t)L * CDIM * 4);
  _Float16* hbuf = (_Float16*)alloc((size_t)L * HIDDEN * 2);
  _Float16* Wqt  = (_Float16*)alloc((size_t)CDIM * CDIM * 2);
  _Float16* Wkt  = (_Float16*)alloc((size_t)CDIM * CDIM * 2);
  _Float16* Wvt  = (_Float16*)alloc((size_t)CDIM * CDIM * 2);
  _Float16* Wot  = (_Float16*)alloc((size_t)CDIM * CDIM * 2);
  _Float16* W1t  = (_Float16*)alloc((size_t)HIDDEN * CDIM * 2);
  _Float16* W2t  = (_Float16*)alloc((size_t)CDIM * HIDDEN * 2);

  const int LC = L * CDIM, LH = L * HEADS;

  // ---- weights -> K-major f16 (once per launch; deterministic) ----
  int wn = CDIM * CDIM;
  w_transpose_f16_kernel<<<(wn + 255) / 256, 256, 0, stream>>>(Wq, Wqt, CDIM, CDIM);
  w_transpose_f16_kernel<<<(wn + 255) / 256, 256, 0, stream>>>(Wk, Wkt, CDIM, CDIM);
  w_transpose_f16_kernel<<<(wn + 255) / 256, 256, 0, stream>>>(Wv, Wvt, CDIM, CDIM);
  w_transpose_f16_kernel<<<(wn + 255) / 256, 256, 0, stream>>>(Wo, Wot, CDIM, CDIM);
  int w1n = HIDDEN * CDIM;
  w_transpose_f16_kernel<<<(w1n + 255) / 256, 256, 0, stream>>>(W1, W1t, HIDDEN, CDIM);
  w_transpose_f16_kernel<<<(w1n + 255) / 256, 256, 0, stream>>>(W2, W2t, CDIM, HIDDEN);

  // ---- attention block ----
  ln_to_f16_kernel<<<L, 256, 0, stream>>>(x, lag, lab, zh);

  dim3 gC((CDIM + 63) / 64, (L + 63) / 64);
  gemm_f16_wmma_kernel<0><<<gC, 128, 0, stream>>>(zh, Wqt, bq, nullptr, q,    nullptr, L, CDIM, CDIM);
  gemm_f16_wmma_kernel<0><<<gC, 128, 0, stream>>>(zh, Wkt, bk, nullptr, kbuf, nullptr, L, CDIM, CDIM);
  gemm_f16_wmma_kernel<0><<<gC, 128, 0, stream>>>(zh, Wvt, bv, nullptr, vbuf, nullptr, L, CDIM, CDIM);

  attn_init_kernel<<<(LC + 255) / 256, 256, 0, stream>>>(mmax, den, agg, LH, LC);
  edge_score_kernel<<<(E + 7) / 8, 256, 0, stream>>>(q, kbuf, ei, ab, s, mmax, E);
  edge_p_kernel<<<(E * HEADS + 255) / 256, 256, 0, stream>>>(s, ei, mmax, p, den, E);
  edge_agg_kernel<<<(E + 7) / 8, 256, 0, stream>>>(p, den, vbuf, ei, agg, E);

  f32_to_f16_kernel<<<(LC + 255) / 256, 256, 0, stream>>>(agg, zh, LC);
  // x1 = x + agg @ Wo^T + bo  -> d_out
  gemm_f16_wmma_kernel<1><<<gC, 128, 0, stream>>>(zh, Wot, bo, x, out, nullptr, L, CDIM, CDIM);

  // ---- feed-forward block ----
  ln_to_f16_kernel<<<L, 256, 0, stream>>>(out, lfg, lfb, zh);
  dim3 gH((HIDDEN + 63) / 64, (L + 63) / 64);
  gemm_f16_wmma_kernel<2><<<gH, 128, 0, stream>>>(zh, W1t, b1, nullptr, nullptr, hbuf, L, HIDDEN, CDIM);
  // out = x1 + silu(h) @ W2^T + b2  (in-place residual: each element read+written by one thread)
  gemm_f16_wmma_kernel<1><<<gC, 128, 0, stream>>>(hbuf, W2t, b2, out, out, nullptr, L, CDIM, HIDDEN);
}
